// CompressedLinear_17222818857040
// MI455X (gfx1250) — matrix operand order repaired and checked by reading, not compiled, not run on hardware
//
#include <hip/hip_runtime.h>

typedef __attribute__((ext_vector_type(16))) _Float16 v16h;
typedef __attribute__((ext_vector_type(8)))  _Float16 v8h;
typedef __attribute__((ext_vector_type(4)))  _Float16 v4h;
typedef __attribute__((ext_vector_type(8)))  float    v8f;

#define M_DIM 8192      // B * S
#define N_DIM 16384     // OUT
#define K_DIM 4096      // IN
#define BM 128
#define BN 128
#define BK 32
#define KT (K_DIM / BK)     // 128 k-steps
#define LDSS 40             // LDS row stride in halves (32 + 8 pad -> conflict-free b128 reads)

__launch_bounds__(256)
__global__ void qlinear_wmma_f16(const float* __restrict__ x,
                                 const int*   __restrict__ w,
                                 const float* __restrict__ scale,
                                 const float* __restrict__ bias,
                                 float*       __restrict__ out)
{
    __shared__ _Float16 As[2][BM * LDSS];   // x tile, f16
    __shared__ _Float16 Bs[2][BN * LDSS];   // W tile, f16 (ints in [-64,63] are exact)

    const int tid  = threadIdx.x;
    const int lane = tid & 31;
    const int wave = tid >> 5;
    const int wm   = wave & 3;   // wave row   (4 waves over M)
    const int wn   = wave >> 2;  // wave col   (2 waves over N)
    const int l    = lane & 15;
    const int kh   = lane >> 4;  // half-wave select

    // Block swizzle: 8 N-blocks x 64 M-blocks per group -> W working set 16 MB,
    // x (134 MB) stays resident in the 192 MB L2 across the dispatch.
    const int bid     = blockIdx.x;
    const int group   = bid / (8 * (M_DIM / BM));
    const int inGroup = bid % (8 * (M_DIM / BM));
    const int bn      = group * 8 + (inGroup & 7);
    const int bm      = inGroup >> 3;
    const int rowBase = bm * BM;
    const int colBase = bn * BN;

    // global-load thread mapping: 256 threads cover 32 rows x 32 k per pass, 4 passes
    const int lr = tid >> 3;        // 0..31 row-in-pass
    const int lc = (tid & 7) * 4;   // 0..28 k offset (float4 / int4)

    float4 aR[4];
    int4   bR[4];

    // epilogue constants
    float sc[4], bi[4];
#pragma unroll
    for (int tn = 0; tn < 4; ++tn) {
        const int o = colBase + wn * 64 + tn * 16 + l;
        sc[tn] = scale[o];
        bi[tn] = bias[o];
    }

    const v8f vzero = {0.f, 0.f, 0.f, 0.f, 0.f, 0.f, 0.f, 0.f};
    v8f acc[2][4];
#pragma unroll
    for (int tm = 0; tm < 2; ++tm)
#pragma unroll
        for (int tn = 0; tn < 4; ++tn)
            acc[tm][tn] = vzero;

    // ---- prologue: stage k-step 0 into LDS buffer 0 ----
#pragma unroll
    for (int p = 0; p < 4; ++p) {
        const int r = p * 32 + lr;
        aR[p] = *(const float4*)(x + (size_t)(rowBase + r) * K_DIM + lc);
        bR[p] = *(const int4*)  (w + (size_t)(colBase + r) * K_DIM + lc);
    }
#pragma unroll
    for (int p = 0; p < 4; ++p) {
        const int r = p * 32 + lr;
        v4h ha = {(_Float16)aR[p].x, (_Float16)aR[p].y, (_Float16)aR[p].z, (_Float16)aR[p].w};
        *(v4h*)(&As[0][r * LDSS + lc]) = ha;
        v4h hb = {(_Float16)bR[p].x, (_Float16)bR[p].y, (_Float16)bR[p].z, (_Float16)bR[p].w};
        *(v4h*)(&Bs[0][r * LDSS + lc]) = hb;
    }

    int buf = 0;
    for (int kt = 0; kt < KT; ++kt) {
        __syncthreads();

        const bool more = (kt + 1) < KT;
        if (more) {
            const int kc = (kt + 1) * BK + lc;
#pragma unroll
            for (int p = 0; p < 4; ++p) {
                const int r = p * 32 + lr;
                aR[p] = *(const float4*)(x + (size_t)(rowBase + r) * K_DIM + kc);
                bR[p] = *(const int4*)  (w + (size_t)(colBase + r) * K_DIM + kc);
            }
        }

        const _Float16* Ab = As[buf];
        const _Float16* Bb = Bs[buf];

        // A fragment (16x32 f16): lanes 0-15 hold K 0..7 & 16..23, lanes 16-31 hold K 8..15 & 24..31
        v16h a[2];
#pragma unroll
        for (int tm = 0; tm < 2; ++tm) {
            const _Float16* ap = Ab + (wm * 32 + tm * 16 + l) * LDSS + kh * 8;
            v8h alo = *(const v8h*)(ap);
            v8h ahi = *(const v8h*)(ap + 16);
            a[tm] = __builtin_shufflevector(alo, ahi, 0,1,2,3,4,5,6,7,8,9,10,11,12,13,14,15);
        }
        // B fragment (32x16 f16): lane = N, lanes 0-15 hold K 0..7 & 16..23, lanes 16-31 hold K 8..15 & 24..31
        v16h b[4];
#pragma unroll
        for (int tn = 0; tn < 4; ++tn) {
            const _Float16* bp = Bb + (wn * 64 + tn * 16 + l) * LDSS + kh * 8;
            v8h blo = *(const v8h*)(bp);
            v8h bhi = *(const v8h*)(bp + 16);
            b[tn] = __builtin_shufflevector(blo, bhi, 0,1,2,3,4,5,6,7,8,9,10,11,12,13,14,15);
        }

#pragma unroll
        for (int tm = 0; tm < 2; ++tm)
#pragma unroll
            for (int tn = 0; tn < 4; ++tn)
                acc[tm][tn] = __builtin_amdgcn_wmma_f32_16x16x32_f16(
                    false, a[tm], false, b[tn], (short)0, acc[tm][tn], false, false);

        if (more) {
            const int nb = buf ^ 1;
#pragma unroll
            for (int p = 0; p < 4; ++p) {
                const int r = p * 32 + lr;
                v4h ha = {(_Float16)aR[p].x, (_Float16)aR[p].y, (_Float16)aR[p].z, (_Float16)aR[p].w};
                *(v4h*)(&As[nb][r * LDSS + lc]) = ha;
                v4h hb = {(_Float16)bR[p].x, (_Float16)bR[p].y, (_Float16)bR[p].z, (_Float16)bR[p].w};
                *(v4h*)(&Bs[nb][r * LDSS + lc]) = hb;
            }
        }
        buf ^= 1;
    }

    // ---- epilogue: dequant scale + bias, streamed non-temporal stores ----
#pragma unroll
    for (int tm = 0; tm < 2; ++tm) {
        const int m0 = rowBase + wm * 32 + tm * 16 + kh * 8;
#pragma unroll
        for (int tn = 0; tn < 4; ++tn) {
            const int o = colBase + wn * 64 + tn * 16 + l;
            float* op = out + (size_t)m0 * N_DIM + o;
            const float s = sc[tn];
            const float bv = bi[tn];
            v8f c = acc[tm][tn];
#pragma unroll
            for (int j = 0; j < 8; ++j)
                __builtin_nontemporal_store(c[j] * s + bv, op + (size_t)j * N_DIM);
        }
    }
}

extern "C" void kernel_launch(void* const* d_in, const int* in_sizes, int n_in,
                              void* d_out, int out_size, void* d_ws, size_t ws_size,
                              hipStream_t stream) {
    (void)in_sizes; (void)n_in; (void)out_size; (void)d_ws; (void)ws_size;
    const float* x     = (const float*)d_in[0];
    const int*   w     = (const int*)d_in[1];
    const float* scale = (const float*)d_in[2];
    const float* bias  = (const float*)d_in[3];
    float*       out   = (float*)d_out;

    const int grid = (M_DIM / BM) * (N_DIM / BN);   // 64 * 128 = 8192 workgroups
    qlinear_wmma_f16<<<grid, 256, 0, stream>>>(x, w, scale, bias, out);
}
